// CMB_17858474017139
// MI455X (gfx1250) — matrix operand order repaired and checked by reading, not compiled, not run on hardware
//
#include <hip/hip_runtime.h>
#include <hip/hip_bf16.h>
#include <stdint.h>

// ---------------------------------------------------------------------------
// Shapes: B=32768, A=32, D=64, H=128, V=50000
//   hidden = ctx(16x128) @ W1[a]^T        -> 8 Ntiles x 4 Ktiles wmma / head
//   preds  = hidden(16x128) @ W2[a]^T     -> 4 x 4 wmma / head
//   h2     = relu(hctx + preds(16x64)@Wp^T + br1) -> 8 x 2 wmma / head
//   hctx   = ctx(16x128) @ Wc^T           -> 8 Ntiles (1 per wave) x 4 K
// All matmuls via v_wmma_f32_16x16x32_bf16 (bf16 in, f32 accumulate).
// ---------------------------------------------------------------------------

typedef __attribute__((ext_vector_type(8)))  float    v8f;
typedef __attribute__((ext_vector_type(16))) __bf16   v16bf;
typedef __attribute__((ext_vector_type(4)))  uint32_t u32x4;

union FragAB {
  v16bf v;
  u32x4 q[2];   // two 16-byte halves (8 bf16 each)
};

// bf16-element offsets inside d_ws for the repacked (B-fragment order) weights
#define WS_W1 0          // 32 heads * 32 tiles * 512 el  = 524288
#define WS_W2 524288     // 32 heads * 16 tiles * 512 el  = 262144
#define WS_WC 786432     // 32 tiles * 512 el             = 16384
#define WS_WP 802816     // 16 tiles * 512 el             = 8192

static __device__ inline v8f wmma_bf16(const FragAB& a, const FragAB& b, v8f c) {
  return __builtin_amdgcn_wmma_f32_16x16x32_bf16(
      /*neg_a=*/false, a.v, /*neg_b=*/false, b.v,
      /*c_mod=*/(short)0, c, /*reuse_a=*/false, /*reuse_b=*/false);
}

// ---------------------------------------------------------------------------
// Repack f32 weights -> bf16 WMMA-B-fragment layout in d_ws.
// One "chunk" = one lane's 16 contiguous-K bf16 values (32 bytes).
// B fragment (K=32,N=16): lane l -> n = l&15, K block = (l>>4)*16 .. +15.
// ---------------------------------------------------------------------------
__global__ void repack_weights(const float* __restrict__ W1,
                               const float* __restrict__ W2,
                               const float* __restrict__ Wr1,
                               __bf16* __restrict__ ws) {
  int c = blockIdx.x * blockDim.x + threadIdx.x;
  const float* src;
  __bf16* dst;
  if (c < 32768) {                                  // W1 (A,H,2D)
    int tile = c >> 5, l = c & 31;
    int a = tile >> 5, rem = tile & 31;
    int nt = rem >> 2, kt = rem & 3;
    int n = l & 15, hf = l >> 4;
    src = W1 + a * (128 * 128) + (nt * 16 + n) * 128 + kt * 32 + hf * 16;
    dst = ws + WS_W1 + c * 16;
  } else if (c < 32768 + 16384) {                   // W2 (A,D,H)
    int c2 = c - 32768;
    int tile = c2 >> 5, l = c2 & 31;
    int a = tile >> 4, rem = tile & 15;
    int nt = rem >> 2, kt = rem & 3;
    int n = l & 15, hf = l >> 4;
    src = W2 + a * (64 * 128) + (nt * 16 + n) * 128 + kt * 32 + hf * 16;
    dst = ws + WS_W2 + c2 * 16;
  } else if (c < 32768 + 16384 + 1024) {            // Wc = Wr1[:, :128]
    int c3 = c - (32768 + 16384);
    int tile = c3 >> 5, l = c3 & 31;
    int nt = tile >> 2, kt = tile & 3;
    int n = l & 15, hf = l >> 4;
    src = Wr1 + (nt * 16 + n) * 192 + kt * 32 + hf * 16;
    dst = ws + WS_WC + c3 * 16;
  } else if (c < 32768 + 16384 + 1024 + 512) {      // Wp = Wr1[:, 128:192]
    int c4 = c - (32768 + 16384 + 1024);
    int tile = c4 >> 5, l = c4 & 31;
    int nt = tile >> 1, kt = tile & 1;
    int n = l & 15, hf = l >> 4;
    src = Wr1 + (nt * 16 + n) * 192 + 128 + kt * 32 + hf * 16;
    dst = ws + WS_WP + c4 * 16;
  } else {
    return;
  }
#pragma unroll
  for (int i = 0; i < 16; ++i) dst[i] = (__bf16)src[i];
}

// ---------------------------------------------------------------------------
// Fused per-16-row kernel: embeddings -> 4 GEMMs -> rewards -> argmin -> out.
// 256 threads = 8 waves; wave w owns heads [4w, 4w+4).
// ---------------------------------------------------------------------------
__global__ __launch_bounds__(256) void fused_policy(
    const int* __restrict__ x, const float* __restrict__ cemb,
    const float* __restrict__ b1, const float* __restrict__ b2,
    const float* __restrict__ br1, const float* __restrict__ Wr2,
    const __bf16* __restrict__ ws, float* __restrict__ out_r) {

  __shared__ __bf16 ctx_lds[16 * 128];      // ctx bf16, row-major
  __shared__ float  hctx_lds[16 * 128];     // ctx @ Wc^T, f32
  __shared__ __bf16 hid_lds[8][16 * 128];   // per-wave hidden staging (bf16)
  __shared__ __bf16 pst_lds[8][16 * 64];    // per-wave preds staging (bf16)
  __shared__ float  rewards_lds[32 * 16];   // [head][row]
  __shared__ int    besta[16];

  const int tid   = threadIdx.x;
  const int wave  = tid >> 5;
  const int lane  = tid & 31;
  const int nlane = lane & 15;
  const int hf    = lane >> 4;              // lane half (C-layout M offset 0/8)
  const int b0    = blockIdx.x * 16;

  // ---- Phase 0: gather ctx embeddings -> bf16 LDS (16 x 128 row-major) ----
  {
    int row  = tid >> 4;
    int part = tid & 15;
    int c0   = part * 8;
    int which = (c0 >= 64);
    int idx = x[(b0 + row) * 2 + which];
    const float* src = cemb + idx * 64 + (c0 - which * 64);
#pragma unroll
    for (int i = 0; i < 8; ++i)
      ctx_lds[row * 128 + c0 + i] = (__bf16)src[i];
  }
  __syncthreads();

  // ---- A fragments of ctx (K tiles 0..3), reused by every GEMM on ctx ----
  // 16-bit A layout: lane holds row m=lane&15; K = kt*32 + hf*16 .. +15 (contiguous)
  FragAB actx[4];
#pragma unroll
  for (int kt = 0; kt < 4; ++kt) {
    const __bf16* rp = &ctx_lds[nlane * 128 + kt * 32 + hf * 16];
    actx[kt].q[0] = *(const u32x4*)rp;
    actx[kt].q[1] = *(const u32x4*)(rp + 8);
  }

  // ---- Phase 1: h_ctx — wave w computes N-tile w (h = 16w..16w+15) ----
  {
    v8f c = {};
#pragma unroll
    for (int kt = 0; kt < 4; ++kt) {
      FragAB bw;
      const __bf16* bt = ws + WS_WC + (wave * 4 + kt) * 512 + lane * 16;
      bw.q[0] = *(const u32x4*)bt;
      bw.q[1] = *(const u32x4*)(bt + 8);
      c = wmma_bf16(actx[kt], bw, c);
    }
#pragma unroll
    for (int r = 0; r < 8; ++r)
      hctx_lds[(r + hf * 8) * 128 + wave * 16 + nlane] = c[r];
  }
  __syncthreads();

  // ---- Phase 2: per-head pipeline; preds kept live in registers ----
  v8f pacc[4][4];                           // [head-in-wave][d N-tile]
  const int wbase = wave * 4;

#pragma unroll
  for (int j = 0; j < 4; ++j) {
    const int a = wbase + j;
    __bf16* hid = hid_lds[wave];
    __bf16* pst = pst_lds[wave];

    // hidden(16x128) = ctx @ W1[a]^T, +b1, -> bf16 staging
    for (int nt = 0; nt < 8; ++nt) {
      v8f c = {};
#pragma unroll
      for (int kt = 0; kt < 4; ++kt) {
        FragAB bw;
        const __bf16* bt = ws + WS_W1 + (a * 32 + nt * 4 + kt) * 512 + lane * 16;
        bw.q[0] = *(const u32x4*)bt;
        bw.q[1] = *(const u32x4*)(bt + 8);
        c = wmma_bf16(actx[kt], bw, c);
      }
      float b1v = b1[a * 128 + nt * 16 + nlane];
#pragma unroll
      for (int r = 0; r < 8; ++r)
        hid[(r + hf * 8) * 128 + nt * 16 + nlane] = (__bf16)(c[r] + b1v);
    }

    // preds(16x64) = hidden @ W2[a]^T, +b2 -> f32 regs + bf16 staging
#pragma unroll
    for (int nt = 0; nt < 4; ++nt) {
      v8f c = {};
#pragma unroll
      for (int kt = 0; kt < 4; ++kt) {
        FragAB af, bw;
        const __bf16* rp = &hid[nlane * 128 + kt * 32 + hf * 16];
        af.q[0] = *(const u32x4*)rp;
        af.q[1] = *(const u32x4*)(rp + 8);
        const __bf16* bt = ws + WS_W2 + (a * 16 + nt * 4 + kt) * 512 + lane * 16;
        bw.q[0] = *(const u32x4*)bt;
        bw.q[1] = *(const u32x4*)(bt + 8);
        c = wmma_bf16(af, bw, c);
      }
      float b2v = b2[a * 64 + nt * 16 + nlane];
#pragma unroll
      for (int r = 0; r < 8; ++r) {
        float pv = c[r] + b2v;
        c[r] = pv;
        pst[(r + hf * 8) * 64 + nt * 16 + nlane] = (__bf16)pv;
      }
      pacc[j][nt] = c;
    }

    // h2 = relu(hctx + preds @ Wp^T + br1); rewards = h2 . Wr2
    float racc[8];
#pragma unroll
    for (int r = 0; r < 8; ++r) racc[r] = 0.f;
    for (int nt = 0; nt < 8; ++nt) {
      v8f c = {};
#pragma unroll
      for (int kt = 0; kt < 2; ++kt) {
        FragAB af, bw;
        const __bf16* rp = &pst[nlane * 64 + kt * 32 + hf * 16];
        af.q[0] = *(const u32x4*)rp;
        af.q[1] = *(const u32x4*)(rp + 8);
        const __bf16* bt = ws + WS_WP + (nt * 2 + kt) * 512 + lane * 16;
        bw.q[0] = *(const u32x4*)bt;
        bw.q[1] = *(const u32x4*)(bt + 8);
        c = wmma_bf16(af, bw, c);
      }
      int h = nt * 16 + nlane;
      float br1v = br1[h];
      float wr2v = Wr2[h];
#pragma unroll
      for (int r = 0; r < 8; ++r) {
        float v = c[r] + hctx_lds[(r + hf * 8) * 128 + h] + br1v;
        v = fmaxf(v, 0.f);
        racc[r] += v * wr2v;
      }
    }
    // reduce across the 16 N-lanes of each half (masks 1/2/4/8 stay in-half)
#pragma unroll
    for (int r = 0; r < 8; ++r) {
      float s = racc[r];
      s += __shfl_xor(s, 1);
      s += __shfl_xor(s, 2);
      s += __shfl_xor(s, 4);
      s += __shfl_xor(s, 8);
      if (nlane == 0) rewards_lds[a * 16 + r + hf * 8] = s;  // br2 constant: dropped
    }
  }
  __syncthreads();

  // ---- Phase 3: argmin over heads (first-min, matching jnp.argmin) ----
  if (tid < 16) {
    int best = 0;
    float bv = rewards_lds[tid];
    for (int a = 1; a < 32; ++a) {
      float v = rewards_lds[a * 16 + tid];
      if (v < bv) { bv = v; best = a; }
    }
    besta[tid] = best;
  }
  __syncthreads();

  // ---- Phase 4: write preds of the winning head straight from registers ----
#pragma unroll
  for (int j = 0; j < 4; ++j) {
    int a = wbase + j;
#pragma unroll
    for (int nt = 0; nt < 4; ++nt) {
#pragma unroll
      for (int r = 0; r < 8; ++r) {
        int m = r + hf * 8;
        if (besta[m] == a)
          out_r[(b0 + m) * 64 + nt * 16 + nlane] = pacc[j][nt][r];
      }
    }
  }
}

// ---------------------------------------------------------------------------
// wemb = wemb_table[y]  (pure bandwidth gather, coalesced over D)
// ---------------------------------------------------------------------------
__global__ void gather_wemb(const int* __restrict__ y,
                            const float* __restrict__ wt,
                            float* __restrict__ out) {
  int i = blockIdx.x * 256 + threadIdx.x;
  out[i] = wt[y[i >> 6] * 64 + (i & 63)];
}

extern "C" void kernel_launch(void* const* d_in, const int* in_sizes, int n_in,
                              void* d_out, int out_size, void* d_ws, size_t ws_size,
                              hipStream_t stream) {
  const int*   x    = (const int*)d_in[0];
  const int*   y    = (const int*)d_in[1];
  const float* cemb = (const float*)d_in[2];
  const float* wemb = (const float*)d_in[3];
  const float* W1   = (const float*)d_in[4];
  const float* b1   = (const float*)d_in[5];
  const float* W2   = (const float*)d_in[6];
  const float* b2   = (const float*)d_in[7];
  const float* Wr1  = (const float*)d_in[8];
  const float* br1  = (const float*)d_in[9];
  const float* Wr2  = (const float*)d_in[10];
  // d_in[11] = br2: constant across heads -> cannot change argmin; rewards
  // are not an output, so it is dropped.

  __bf16* ws    = (__bf16*)d_ws;            // 811008 bf16 = ~1.6 MB
  float*  out_r = (float*)d_out;            // (B, D)
  float*  out_w = out_r + 32768 * 64;       // (B, D)

  repack_weights<<<198, 256, 0, stream>>>(W1, W2, Wr1, ws);     // 50688 chunks
  fused_policy  <<<2048, 256, 0, stream>>>(x, cemb, b1, b2, br1, Wr2, ws, out_r);
  gather_wemb   <<<8192, 256, 0, stream>>>(y, wemb, out_w);
}
